// TFAE_55284819034284
// MI455X (gfx1250) — matrix operand order repaired and checked by reading, not compiled, not run on hardware
//
#include <hip/hip_runtime.h>
#include <hip/hip_bf16.h>
#include <math.h>

// ---------------------------------------------------------------------------
// MI455X (gfx1250) implementation of the CNN+Transformer autoencoder.
// GEMMs use v_wmma_f32_16x16x32_f16 (wave32, f32 accumulate) with 32x32
// register blocking per wave (4 WMMA accumulators, A/B fragment reuse);
// fp32 residual stream for LayerNorm; f16 activations/weights for matrix ops.
// ---------------------------------------------------------------------------

typedef __attribute__((ext_vector_type(16))) _Float16 v16h;
typedef __attribute__((ext_vector_type(8)))  _Float16 v8h;
typedef __attribute__((ext_vector_type(8)))  float    v8f;

#define NIMG   4096
#define DM     64
#define NB     128
#define LSEQ   64
#define ROWS   8192          // NB * LSEQ
#define DFFN   4096
#define K3     1568          // 32*7*7  (conv3 / dconv1 GEMM K or N)

#define IMGS_OFF 0
#define BIDX_OFF (NIMG*784)
#define CNT_OFF  (BIDX_OFF + NIMG)
#define IMG_OFF  (CNT_OFF + NB)

static __device__ __forceinline__ float gelu_f(float x) {
  return 0.5f * x * (1.0f + erff(x * 0.70710678118654752440f));
}

union AFrag { v16h v; v8h h[2]; };

#define WMMA_F16(a, b, c) \
  __builtin_amdgcn_wmma_f32_16x16x32_f16(false, (a), false, (b), (short)0, (c), false, false)

// ---------------------------- utility kernels ------------------------------

__global__ void k_zero_f32(float* p, int n) {
  int i = blockIdx.x * blockDim.x + threadIdx.x;
  if (i < n) p[i] = 0.f;
}
__global__ void k_zero_f16(_Float16* p, int n) {
  int i = blockIdx.x * blockDim.x + threadIdx.x;
  if (i < n) p[i] = (_Float16)0.f;
}
__global__ void k_zero_i32(int* p, int n) {
  int i = blockIdx.x * blockDim.x + threadIdx.x;
  if (i < n) p[i] = 0;
}
__global__ void k_f2h(const float* __restrict__ s, _Float16* __restrict__ d, int n) {
  int i = blockIdx.x * blockDim.x + threadIdx.x;
  if (i < n) d[i] = (_Float16)s[i];
}
__global__ void k_i2f(const int* __restrict__ s, float* __restrict__ d, int n) {
  int i = blockIdx.x * blockDim.x + threadIdx.x;
  if (i < n) d[i] = (float)s[i];
}
// dconv1 weight (64,32,7,7) -> [1568,64] f16 with wT[n][c] = w[c*1568+n]
__global__ void k_tcvt_dconv1(const float* __restrict__ w, _Float16* __restrict__ o) {
  int i = blockIdx.x * blockDim.x + threadIdx.x;
  if (i < K3 * DM) { int n = i >> 6, c = i & 63; o[i] = (_Float16)w[c * K3 + n]; }
}
__global__ void k_gather_rows(const float* __restrict__ Z, const int* __restrict__ bidx,
                              _Float16* __restrict__ o) {
  int i = blockIdx.x * blockDim.x + threadIdx.x;
  if (i < NIMG * DM) { int r = i >> 6, d = i & 63; o[i] = (_Float16)Z[(size_t)bidx[r] * DM + d]; }
}
__global__ void k_hist(const int* __restrict__ b, int* __restrict__ c, int n) {
  int i = blockIdx.x * blockDim.x + threadIdx.x;
  if (i < n) atomicAdd(&c[b[i]], 1);
}
__global__ void k_scan128(const int* __restrict__ counts, int* __restrict__ starts,
                          float* __restrict__ cnt_out) {
  __shared__ int sh[NB];
  int t = threadIdx.x;               // 128 threads
  int c = counts[t];
  sh[t] = c;
  __syncthreads();
  for (int o = 1; o < NB; o <<= 1) {
    int v = (t >= o) ? sh[t - o] : 0;
    __syncthreads();
    sh[t] += v;
    __syncthreads();
  }
  starts[t]  = sh[t] - c;            // exclusive prefix
  cnt_out[t] = (float)c;
}
__global__ void k_scatter(const float* __restrict__ feat, const int* __restrict__ bidx,
                          const int* __restrict__ starts, float* __restrict__ S32,
                          _Float16* __restrict__ S16) {
  int i = blockIdx.x * blockDim.x + threadIdx.x;
  if (i >= NIMG * DM) return;
  int r = i >> 6, d = i & 63;
  int b = bidx[r];
  int pos = r - starts[b];
  size_t off = ((size_t)(b * LSEQ + pos)) * DM + d;
  float v = feat[i];
  S32[off] = v;
  S16[off] = (_Float16)v;
}

// ------------------------------- WMMA GEMM ---------------------------------
// Shared epilogue for one 16x16 sub-tile of the C layout:
// VGPR v -> row m0+hi*8+v, col n0+r (r=lane&15, hi=lane>>4).
static __device__ __forceinline__ void store_tile(
    const v8f& acc, int m0, int n0, int N, const float* bias, int bdiv, int act,
    float* Cf, _Float16* Ch, int r, int hi) {
  int n = n0 + r;
  int m = m0 + hi * 8;
  float bv = bias ? bias[n / bdiv] : 0.f;
#pragma unroll
  for (int v = 0; v < 8; v++) {
    float x = acc[v] + bv;
    if (act == 1)      x = fmaxf(x, 0.f);
    else if (act == 2) x = gelu_f(x);
    size_t off = (size_t)(m + v) * N + n;
    if (Cf) Cf[off] = x;
    if (Ch) Ch[off] = (_Float16)x;
  }
}

// C[M,N] = A[M,K](f16,row) * W[N,K](f16,row)^T + bias ; optional relu/gelu.
// One wave computes a 32x32 tile: 4 accumulators, A/B fragments reused 2x.
__global__ void __launch_bounds__(256)
k_gemm(const _Float16* __restrict__ A, const _Float16* __restrict__ W,
       const float* __restrict__ bias, float* __restrict__ Cf,
       _Float16* __restrict__ Ch, int M, int N, int K, int act, int bdiv) {
  int wid = blockIdx.x * 8 + (threadIdx.x >> 5);
  int tn_n = N >> 5;
  int tiles = (M >> 5) * tn_n;
  if (wid >= tiles) return;
  int tm = wid / tn_n, tn = wid - tm * tn_n;
  int lane = threadIdx.x & 31, r = lane & 15, hi = lane >> 4;
  const _Float16* Ap0 = A + (size_t)(tm * 32 + r) * K + hi * 16;
  const _Float16* Ap1 = Ap0 + (size_t)16 * K;
  const _Float16* Wp0 = W + (size_t)(tn * 32 + r) * K + hi * 16;
  const _Float16* Wp1 = Wp0 + (size_t)16 * K;
  v8f acc00 = {}, acc01 = {}, acc10 = {}, acc11 = {};
  for (int k0 = 0; k0 < K; k0 += 32) {
    if (k0 + 32 < K) {                         // gfx1250 global_prefetch_b8
      __builtin_prefetch(Ap0 + k0 + 32, 0, 3);
      __builtin_prefetch(Ap1 + k0 + 32, 0, 3);
      __builtin_prefetch(Wp0 + k0 + 32, 0, 3);
      __builtin_prefetch(Wp1 + k0 + 32, 0, 3);
    }
    AFrag a0, a1, b0, b1;
    a0.v    = *(const v16h*)(Ap0 + k0);
    a1.v    = *(const v16h*)(Ap1 + k0);
    b0.v    = *(const v16h*)(Wp0 + k0);
    b1.v    = *(const v16h*)(Wp1 + k0);
    acc00 = WMMA_F16(a0.v, b0.v, acc00);
    acc01 = WMMA_F16(a0.v, b1.v, acc01);
    acc10 = WMMA_F16(a1.v, b0.v, acc10);
    acc11 = WMMA_F16(a1.v, b1.v, acc11);
  }
  int m0 = tm * 32, n0 = tn * 32;
  store_tile(acc00, m0,      n0,      N, bias, bdiv, act, Cf, Ch, r, hi);
  store_tile(acc01, m0,      n0 + 16, N, bias, bdiv, act, Cf, Ch, r, hi);
  store_tile(acc10, m0 + 16, n0,      N, bias, bdiv, act, Cf, Ch, r, hi);
  store_tile(acc11, m0 + 16, n0 + 16, N, bias, bdiv, act, Cf, Ch, r, hi);
}

// V projection with transposed f16 epilogue: VT[b][d][l] = (A*W^T+bias)[b*64+l][d]
__global__ void __launch_bounds__(256)
k_gemm_vt(const _Float16* __restrict__ A, const _Float16* __restrict__ W,
          const float* __restrict__ bias, _Float16* __restrict__ VT, int M) {
  int wid = blockIdx.x * 8 + (threadIdx.x >> 5);
  int tiles = (M >> 4) * 4;
  if (wid >= tiles) return;
  int tm = wid >> 2, tn = wid & 3;
  int lane = threadIdx.x & 31, r = lane & 15, hi = lane >> 4;
  const _Float16* Ap = A + (size_t)(tm * 16 + r) * DM + hi * 16;
  const _Float16* Wp = W + (size_t)(tn * 16 + r) * DM + hi * 16;
  v8f acc = {};
  for (int k0 = 0; k0 < DM; k0 += 32) {
    AFrag a, b;
    a.v    = *(const v16h*)(Ap + k0);
    b.v    = *(const v16h*)(Wp + k0);
    acc = WMMA_F16(a.v, b.v, acc);
  }
  int d = tn * 16 + r;
  int m0 = tm * 16 + hi * 8;
  float bv = bias[d];
#pragma unroll
  for (int v = 0; v < 8; v++) {
    int m = m0 + v;
    VT[(size_t)(m >> 6) * (LSEQ * DM) + d * LSEQ + (m & 63)] = (_Float16)(acc[v] + bv);
  }
}

// per-batch S[b] = Q_b * K_b^T  (64x64x64, f32 out; softmax applies 1/8 scale)
// one wave computes a 32x32 tile; 4 waves per batch.
__global__ void __launch_bounds__(256)
k_attn_scores(const _Float16* __restrict__ Q, const _Float16* __restrict__ Kp,
              float* __restrict__ S) {
  int wid = blockIdx.x * 8 + (threadIdx.x >> 5);
  if (wid >= NB * 4) return;
  int b = wid >> 2, t = wid & 3, tm = t >> 1, tn = t & 1;
  int lane = threadIdx.x & 31, r = lane & 15, hi = lane >> 4;
  const _Float16* Ap0 = Q + (size_t)(b * LSEQ + tm * 32 + r) * DM + hi * 16;
  const _Float16* Ap1 = Ap0 + (size_t)16 * DM;
  const _Float16* Bp0 = Kp + (size_t)(b * LSEQ + tn * 32 + r) * DM + hi * 16;
  const _Float16* Bp1 = Bp0 + (size_t)16 * DM;
  v8f acc00 = {}, acc01 = {}, acc10 = {}, acc11 = {};
  for (int k0 = 0; k0 < DM; k0 += 32) {
    AFrag a0, a1, b0, b1;
    a0.v    = *(const v16h*)(Ap0 + k0);
    a1.v    = *(const v16h*)(Ap1 + k0);
    b0.v    = *(const v16h*)(Bp0 + k0);
    b1.v    = *(const v16h*)(Bp1 + k0);
    acc00 = WMMA_F16(a0.v, b0.v, acc00);
    acc01 = WMMA_F16(a0.v, b1.v, acc01);
    acc10 = WMMA_F16(a1.v, b0.v, acc10);
    acc11 = WMMA_F16(a1.v, b1.v, acc11);
  }
  float* out = S + (size_t)b * LSEQ * LSEQ;
  int m0 = tm * 32 + hi * 8, n0 = tn * 32 + r;
#pragma unroll
  for (int v = 0; v < 8; v++) {
    out[(m0 + v) * LSEQ + n0]           = acc00[v];
    out[(m0 + v) * LSEQ + n0 + 16]      = acc01[v];
    out[(m0 + 16 + v) * LSEQ + n0]      = acc10[v];
    out[(m0 + 16 + v) * LSEQ + n0 + 16] = acc11[v];
  }
}

// per-batch AO[b*64+l][d] = P_b[l,:] * V_b[:,d]  using VT (row-major [d][l])
__global__ void __launch_bounds__(256)
k_attn_av(const _Float16* __restrict__ P, const _Float16* __restrict__ VT,
          _Float16* __restrict__ AO) {
  int wid = blockIdx.x * 8 + (threadIdx.x >> 5);
  if (wid >= NB * 4) return;
  int b = wid >> 2, t = wid & 3, tm = t >> 1, tn = t & 1;
  int lane = threadIdx.x & 31, r = lane & 15, hi = lane >> 4;
  const _Float16* Ap0 = P  + (size_t)b * LSEQ * LSEQ + (tm * 32 + r) * LSEQ + hi * 16;
  const _Float16* Ap1 = Ap0 + (size_t)16 * LSEQ;
  const _Float16* Bp0 = VT + (size_t)b * LSEQ * DM + (tn * 32 + r) * LSEQ + hi * 16;
  const _Float16* Bp1 = Bp0 + (size_t)16 * LSEQ;
  v8f acc00 = {}, acc01 = {}, acc10 = {}, acc11 = {};
  for (int k0 = 0; k0 < LSEQ; k0 += 32) {
    AFrag a0, a1, b0, b1;
    a0.v    = *(const v16h*)(Ap0 + k0);
    a1.v    = *(const v16h*)(Ap1 + k0);
    b0.v    = *(const v16h*)(Bp0 + k0);
    b1.v    = *(const v16h*)(Bp1 + k0);
    acc00 = WMMA_F16(a0.v, b0.v, acc00);
    acc01 = WMMA_F16(a0.v, b1.v, acc01);
    acc10 = WMMA_F16(a1.v, b0.v, acc10);
    acc11 = WMMA_F16(a1.v, b1.v, acc11);
  }
  _Float16* out = AO + (size_t)b * LSEQ * DM;
  int m0 = tm * 32 + hi * 8, n0 = tn * 32 + r;
#pragma unroll
  for (int v = 0; v < 8; v++) {
    out[(m0 + v) * DM + n0]           = (_Float16)acc00[v];
    out[(m0 + v) * DM + n0 + 16]      = (_Float16)acc01[v];
    out[(m0 + 16 + v) * DM + n0]      = (_Float16)acc10[v];
    out[(m0 + 16 + v) * DM + n0 + 16] = (_Float16)acc11[v];
  }
}

// ------------------------- softmax / layernorm -----------------------------

__global__ void __launch_bounds__(256)
k_softmax(const float* __restrict__ S, _Float16* __restrict__ P) {
  int row = blockIdx.x * 8 + (threadIdx.x >> 5);
  if (row >= ROWS) return;
  int lane = threadIdx.x & 31;
  const float* s = S + (size_t)row * LSEQ;
  float a = s[lane] * 0.125f, b = s[lane + 32] * 0.125f;   // 1/sqrt(64)
  float m = fmaxf(a, b);
  for (int o = 16; o; o >>= 1) m = fmaxf(m, __shfl_xor(m, o, 32));
  float e0 = expf(a - m), e1 = expf(b - m);
  float su = e0 + e1;
  for (int o = 16; o; o >>= 1) su += __shfl_xor(su, o, 32);
  float inv = 1.f / su;
  P[(size_t)row * LSEQ + lane]      = (_Float16)(e0 * inv);
  P[(size_t)row * LSEQ + lane + 32] = (_Float16)(e1 * inv);
}

// Out = LN(X (+Y)) * g + be ; optional f32 and f16 outputs
__global__ void __launch_bounds__(256)
k_add_ln(const float* __restrict__ X, const float* __restrict__ Y,
         const float* __restrict__ g, const float* __restrict__ be,
         float* __restrict__ Of, _Float16* __restrict__ Oh, int rows) {
  int row = blockIdx.x * 8 + (threadIdx.x >> 5);
  if (row >= rows) return;
  int lane = threadIdx.x & 31;
  size_t base = (size_t)row * DM;
  float a = X[base + lane], b = X[base + lane + 32];
  if (Y) { a += Y[base + lane]; b += Y[base + lane + 32]; }
  float s = a + b;
  for (int o = 16; o; o >>= 1) s += __shfl_xor(s, o, 32);
  float mean = s * (1.f / 64.f);
  float da = a - mean, db = b - mean;
  float vs = da * da + db * db;
  for (int o = 16; o; o >>= 1) vs += __shfl_xor(vs, o, 32);
  float rinv = rsqrtf(vs * (1.f / 64.f) + 1e-5f);
  float o0 = da * rinv * g[lane] + be[lane];
  float o1 = db * rinv * g[lane + 32] + be[lane + 32];
  if (Of) { Of[base + lane] = o0; Of[base + lane + 32] = o1; }
  if (Oh) { Oh[base + lane] = (_Float16)o0; Oh[base + lane + 32] = (_Float16)o1; }
}

// ------------------------------ CNN kernels --------------------------------

__global__ void k_conv1(const float* __restrict__ X, const float* __restrict__ W,
                        const float* __restrict__ B, float* __restrict__ O) {
  int id = blockIdx.x * blockDim.x + threadIdx.x;
  if (id >= NIMG * 16 * 196) return;
  int ox = id % 14, oy = (id / 14) % 14, oc = (id / 196) % 16, n = id / (196 * 16);
  const float* x = X + (size_t)n * 784;
  const float* w = W + oc * 9;
  float acc = B[oc];
  for (int ky = 0; ky < 3; ky++) {
    int iy = oy * 2 - 1 + ky; if (iy < 0 || iy >= 28) continue;
    for (int kx = 0; kx < 3; kx++) {
      int ix = ox * 2 - 1 + kx; if (ix < 0 || ix >= 28) continue;
      acc += x[iy * 28 + ix] * w[ky * 3 + kx];
    }
  }
  O[id] = gelu_f(acc);
}

__global__ void k_conv2(const float* __restrict__ H1, const float* __restrict__ W,
                        const float* __restrict__ B, _Float16* __restrict__ O) {
  int id = blockIdx.x * blockDim.x + threadIdx.x;
  if (id >= NIMG * 32 * 49) return;
  int ox = id % 7, oy = (id / 7) % 7, oc = (id / 49) % 32, n = id / (49 * 32);
  const float* x = H1 + (size_t)n * 16 * 196;
  const float* w = W + oc * 144;
  float acc = B[oc];
  for (int ic = 0; ic < 16; ic++) {
    const float* xc = x + ic * 196;
    const float* wc = w + ic * 9;
    for (int ky = 0; ky < 3; ky++) {
      int iy = oy * 2 - 1 + ky; if (iy < 0 || iy >= 14) continue;
      for (int kx = 0; kx < 3; kx++) {
        int ix = ox * 2 - 1 + kx; if (ix < 0 || ix >= 14) continue;
        acc += xc[iy * 14 + ix] * wc[ky * 3 + kx];
      }
    }
  }
  O[id] = (_Float16)gelu_f(acc);   // feeds conv3 WMMA GEMM (K=1568)
}

// ConvTranspose2d(32->16, k3, s2, p1, op1), 7->14, gelu
__global__ void k_dconv2(const float* __restrict__ G1, const float* __restrict__ W,
                         const float* __restrict__ B, float* __restrict__ O) {
  int id = blockIdx.x * blockDim.x + threadIdx.x;
  if (id >= NIMG * 16 * 196) return;
  int ox = id % 14, oy = (id / 14) % 14, oc = (id / 196) % 16, n = id / (196 * 16);
  const float* x = G1 + (size_t)n * 32 * 49;
  float acc = B[oc];
  for (int ic = 0; ic < 32; ic++) {
    const float* xc = x + ic * 49;
    const float* wc = W + (ic * 16 + oc) * 9;
    for (int ky = 0; ky < 3; ky++) {
      int t = oy + 1 - ky; if (t & 1) continue;
      int iy = t >> 1; if (iy < 0 || iy > 6) continue;
      for (int kx = 0; kx < 3; kx++) {
        int s = ox + 1 - kx; if (s & 1) continue;
        int ix = s >> 1; if (ix < 0 || ix > 6) continue;
        acc += xc[iy * 7 + ix] * wc[ky * 3 + kx];
      }
    }
  }
  O[id] = gelu_f(acc);
}

// ConvTranspose2d(16->1, k3, s2, p1, op1), 14->28
__global__ void k_dconv3(const float* __restrict__ X, const float* __restrict__ W,
                         const float* __restrict__ B, float* __restrict__ O) {
  int id = blockIdx.x * blockDim.x + threadIdx.x;
  if (id >= NIMG * 784) return;
  int ox = id % 28, oy = (id / 28) % 28, n = id / 784;
  float acc = B[0];
  for (int ic = 0; ic < 16; ic++) {
    const float* xc = X + ((size_t)n * 16 + ic) * 196;
    const float* wc = W + ic * 9;
    for (int ky = 0; ky < 3; ky++) {
      int t = oy + 1 - ky; if (t & 1) continue;
      int iy = t >> 1; if (iy < 0 || iy > 13) continue;
      for (int kx = 0; kx < 3; kx++) {
        int s = ox + 1 - kx; if (s & 1) continue;
        int ix = s >> 1; if (ix < 0 || ix > 13) continue;
        acc += xc[iy * 14 + ix] * wc[ky * 3 + kx];
      }
    }
  }
  O[id] = acc;
}

// ------------------------------- host side ---------------------------------
// Input index map (dict insertion order):
//  0:x 1:b_idx  2..13: conv1..dconv3 (w,b pairs)
//  enc layer i base E=14+16i: wq.w,wq.b,wk.w,wk.b,wv.w,wv.b,wo.w,wo.b,
//                             lin1.w,lin1.b,lin2.w,lin2.b,ln1.g,ln1.b,ln2.g,ln2.b
//  dec layer i base D=110+26i: sa(8), ca(8), lin1(2), lin2(2), ln1..ln3(6)
//  266,267: enc_norm g,b   268,269: dec_norm g,b

extern "C" void kernel_launch(void* const* d_in, const int* in_sizes, int n_in,
                              void* d_out, int out_size, void* d_ws, size_t ws_size,
                              hipStream_t stream) {
  (void)in_sizes; (void)n_in; (void)out_size; (void)ws_size;
  auto FP = [&](int i) { return (const float*)d_in[i]; };
  const int* BIDX = (const int*)d_in[1];
  float* outF = (float*)d_out;

  // ---- workspace bump allocator (256B aligned) ----
  char* wsp = (char*)d_ws;
  auto alloc = [&](size_t bytes) { char* p = wsp; wsp += (bytes + 255) & ~(size_t)255; return p; };

  int*      cnt_i  = (int*)alloc(NB * 4);
  int*      strt_i = (int*)alloc(NB * 4);
  _Float16* wq16   = (_Float16*)alloc(DM * DM * 2);
  _Float16* wk16   = (_Float16*)alloc(DM * DM * 2);
  _Float16* wv16   = (_Float16*)alloc(DM * DM * 2);
  _Float16* wo16   = (_Float16*)alloc(DM * DM * 2);
  _Float16* w116   = (_Float16*)alloc((size_t)DFFN * DM * 2);
  _Float16* w216   = (_Float16*)alloc((size_t)DM * DFFN * 2);
  _Float16* c3w16  = (_Float16*)alloc((size_t)DM * K3 * 2);
  _Float16* d1wT16 = (_Float16*)alloc((size_t)K3 * DM * 2);
  float*    h1     = (float*)alloc((size_t)NIMG * 16 * 196 * 4);   // conv1 out / dconv2 out
  _Float16* h2h    = (_Float16*)alloc((size_t)NIMG * K3 * 2);      // conv2 out (f16)
  float*    feat32 = (float*)alloc((size_t)NIMG * DM * 4);
  _Float16* feat16 = (_Float16*)alloc((size_t)NIMG * DM * 2);
  float*    src32  = (float*)alloc((size_t)ROWS * DM * 4);
  _Float16* src16  = (_Float16*)alloc((size_t)ROWS * DM * 2);
  float*    mem32  = (float*)alloc((size_t)ROWS * DM * 4);
  _Float16* mem16  = (_Float16*)alloc((size_t)ROWS * DM * 2);
  float*    tgt32  = (float*)alloc((size_t)ROWS * DM * 4);
  _Float16* tgt16  = (_Float16*)alloc((size_t)ROWS * DM * 2);
  _Float16* q16    = (_Float16*)alloc((size_t)ROWS * DM * 2);
  _Float16* kk16   = (_Float16*)alloc((size_t)ROWS * DM * 2);
  _Float16* vt16   = (_Float16*)alloc((size_t)ROWS * DM * 2);
  _Float16* ao16   = (_Float16*)alloc((size_t)ROWS * DM * 2);
  _Float16* p16    = (_Float16*)alloc((size_t)ROWS * LSEQ * 2);
  float*    s32    = (float*)alloc((size_t)ROWS * LSEQ * 4);
  float*    o32    = (float*)alloc((size_t)ROWS * DM * 4);
  _Float16* ffh16  = (_Float16*)alloc((size_t)ROWS * DFFN * 2);
  float*    zbuf32 = (float*)alloc((size_t)ROWS * DM * 4);
  _Float16* zg16   = (_Float16*)alloc((size_t)NIMG * DM * 2);
  float*    g1f32  = (float*)alloc((size_t)NIMG * K3 * 4);

  auto G = [](long n) { return dim3((unsigned)((n + 255) / 256)); };
  auto cvt = [&](const float* s, _Float16* d, int n) {
    k_f2h<<<G(n), dim3(256), 0, stream>>>(s, d, n);
  };
  auto gemm = [&](const _Float16* A, const _Float16* W, const float* bias,
                  float* Cf, _Float16* Ch, int M, int N, int K, int act, int bdiv) {
    long tiles = (long)(M / 32) * (N / 32);          // 32x32 tile per wave
    k_gemm<<<dim3((unsigned)((tiles + 7) / 8)), dim3(256), 0, stream>>>(
        A, W, bias, Cf, Ch, M, N, K, act, bdiv);
  };
  auto addln = [&](const float* X, const float* Y, const float* g, const float* b,
                   float* Of, _Float16* Oh) {
    k_add_ln<<<dim3(ROWS / 8), dim3(256), 0, stream>>>(X, Y, g, b, Of, Oh, ROWS);
  };
  auto attention = [&](const _Float16* qsrc, const _Float16* kvsrc,
                       const _Float16* wq, const float* bq,
                       const _Float16* wk, const float* bk,
                       const _Float16* wv, const float* bv,
                       const _Float16* wo, const float* bo) {
    gemm(qsrc, wq, bq, nullptr, q16, ROWS, DM, DM, 0, 1);
    gemm(kvsrc, wk, bk, nullptr, kk16, ROWS, DM, DM, 0, 1);
    k_gemm_vt<<<dim3(256), dim3(256), 0, stream>>>(kvsrc, wv, bv, vt16, ROWS);
    k_attn_scores<<<dim3(NB * 4 / 8), dim3(256), 0, stream>>>(q16, kk16, s32);
    k_softmax<<<dim3(ROWS / 8), dim3(256), 0, stream>>>(s32, p16);
    k_attn_av<<<dim3(NB * 4 / 8), dim3(256), 0, stream>>>(p16, vt16, ao16);
    gemm(ao16, wo, bo, o32, nullptr, ROWS, DM, DM, 0, 1);
  };

  // ---- one-time weight prep ----
  cvt(FP(6), c3w16, DM * K3);
  k_tcvt_dconv1<<<G(K3 * DM), dim3(256), 0, stream>>>(FP(8), d1wT16);

  // ---- CNN encoder ----
  k_conv1<<<G((long)NIMG * 16 * 196), dim3(256), 0, stream>>>(FP(0), FP(2), FP(3), h1);
  k_conv2<<<G((long)NIMG * 32 * 49), dim3(256), 0, stream>>>(h1, FP(4), FP(5), h2h);
  gemm(h2h, c3w16, FP(7), feat32, feat16, NIMG, DM, K3, 0, 1);

  // ---- ragged pack ----
  k_zero_i32<<<dim3(1), dim3(NB), 0, stream>>>(cnt_i, NB);
  k_hist<<<G(NIMG), dim3(256), 0, stream>>>(BIDX, cnt_i, NIMG);
  k_scan128<<<dim3(1), dim3(NB), 0, stream>>>(cnt_i, strt_i, outF + CNT_OFF);
  k_zero_f32<<<G((long)ROWS * DM), dim3(256), 0, stream>>>(src32, ROWS * DM);
  k_zero_f16<<<G((long)ROWS * DM), dim3(256), 0, stream>>>(src16, ROWS * DM);
  k_scatter<<<G((long)NIMG * DM), dim3(256), 0, stream>>>(feat32, BIDX, strt_i, src32, src16);
  k_i2f<<<G(NIMG), dim3(256), 0, stream>>>(BIDX, outF + BIDX_OFF, NIMG);

  // ---- transformer encoder ----
  for (int L = 0; L < 6; ++L) {
    int E = 14 + 16 * L;
    cvt(FP(E + 0), wq16, DM * DM); cvt(FP(E + 2), wk16, DM * DM);
    cvt(FP(E + 4), wv16, DM * DM); cvt(FP(E + 6), wo16, DM * DM);
    cvt(FP(E + 8), w116, DFFN * DM); cvt(FP(E + 10), w216, DM * DFFN);
    attention(src16, src16, wq16, FP(E + 1), wk16, FP(E + 3), wv16, FP(E + 5), wo16, FP(E + 7));
    addln(src32, o32, FP(E + 12), FP(E + 13), src32, src16);
    gemm(src16, w116, FP(E + 9), nullptr, ffh16, ROWS, DFFN, DM, 1, 1);    // relu
    gemm(ffh16, w216, FP(E + 11), o32, nullptr, ROWS, DM, DFFN, 0, 1);
    addln(src32, o32, FP(E + 14), FP(E + 15), src32, src16);
  }
  addln(src32, nullptr, FP(266), FP(267), mem32, mem16);

  // ---- transformer decoder (tgt starts at zero) ----
  k_zero_f32<<<G((long)ROWS * DM), dim3(256), 0, stream>>>(tgt32, ROWS * DM);
  k_zero_f16<<<G((long)ROWS * DM), dim3(256), 0, stream>>>(tgt16, ROWS * DM);
  for (int L = 0; L < 6; ++L) {
    int D = 110 + 26 * L;
    // self attention
    cvt(FP(D + 0), wq16, DM * DM); cvt(FP(D + 2), wk16, DM * DM);
    cvt(FP(D + 4), wv16, DM * DM); cvt(FP(D + 6), wo16, DM * DM);
    attention(tgt16, tgt16, wq16, FP(D + 1), wk16, FP(D + 3), wv16, FP(D + 5), wo16, FP(D + 7));
    addln(tgt32, o32, FP(D + 20), FP(D + 21), tgt32, tgt16);
    // cross attention (kv from encoder memory)
    cvt(FP(D + 8), wq16, DM * DM); cvt(FP(D + 10), wk16, DM * DM);
    cvt(FP(D + 12), wv16, DM * DM); cvt(FP(D + 14), wo16, DM * DM);
    attention(tgt16, mem16, wq16, FP(D + 9), wk16, FP(D + 11), wv16, FP(D + 13), wo16, FP(D + 15));
    addln(tgt32, o32, FP(D + 22), FP(D + 23), tgt32, tgt16);
    // FFN
    cvt(FP(D + 16), w116, DFFN * DM); cvt(FP(D + 18), w216, DM * DFFN);
    gemm(tgt16, w116, FP(D + 17), nullptr, ffh16, ROWS, DFFN, DM, 1, 1);
    gemm(ffh16, w216, FP(D + 19), o32, nullptr, ROWS, DM, DFFN, 0, 1);
    addln(tgt32, o32, FP(D + 24), FP(D + 25), tgt32, tgt16);
  }
  addln(tgt32, nullptr, FP(268), FP(269), zbuf32, nullptr);

  // z.reshape(-1,64)[b_idx] gather (faithful quirk), f16 for decode GEMM
  k_gather_rows<<<G((long)NIMG * DM), dim3(256), 0, stream>>>(zbuf32, BIDX, zg16);

  // ---- CNN decoder (run twice: raw features -> img, gathered z -> imgs) ----
  auto decode = [&](const _Float16* zin, float* outp) {
    gemm(zin, d1wT16, FP(9), g1f32, nullptr, NIMG, K3, DM, 2, 49);   // dconv1 as GEMM + gelu
    k_dconv2<<<G((long)NIMG * 16 * 196), dim3(256), 0, stream>>>(g1f32, FP(10), FP(11), h1);
    k_dconv3<<<G((long)NIMG * 784), dim3(256), 0, stream>>>(h1, FP(12), FP(13), outp);
  };
  decode(feat16, outF + IMG_OFF);    // img  = cnn_decode(cnn_feat)
  decode(zg16, outF + IMGS_OFF);     // imgs = cnn_decode(z[...])
}
